// MultiHeadAttention_11596411699505
// MI455X (gfx1250) — matrix operand order repaired and checked by reading, not compiled, not run on hardware
//
#include <hip/hip_runtime.h>

// CDNA5 / gfx1250 wave32 WMMA flash attention (causal).
// bf16 WMMA matmuls with f32 softmax/accum; K/V staged once per workgroup as
// packed bf16 in LDS; software-pipelined global loads; diagonal-only causal
// masking; log2-domain softmax (single v_exp_f32 per element); all f32->bf16
// conversion via vector fpround (lowers to two-source v_cvt_pk_bf16_f32).
typedef __attribute__((ext_vector_type(16))) __bf16 v16bf;
typedef __attribute__((ext_vector_type(8)))  __bf16 v8bf;
typedef __attribute__((ext_vector_type(16))) float  v16f;
typedef __attribute__((ext_vector_type(8)))  float  v8f;

#define WMMA_BF16(a, b, c) \
  __builtin_amdgcn_wmma_f32_16x16x32_bf16(false, (a), false, (b), (short)0, (c), false, false)

constexpr int kB = 4, kH = 16, kS = 4096, kD = 64;
constexpr int BLOCK_M = 128;   // queries per workgroup (16 per wave x 8 waves)
constexpr int BLOCK_N = 32;    // keys per iteration (2 x 16-key WMMA subtiles)
constexpr int NTHREADS = 256;
constexpr int KVSTR = 36;      // staged row stride in dwords (32 data + 4 pad, keeps 16B align)

// lane <-> lane^16 exchange (SWAPX16 ds_swizzle, group-of-32 xor mask 0x10)
__device__ __forceinline__ float xor16f(float x) {
  int i = __builtin_amdgcn_ds_swizzle(__builtin_bit_cast(int, x), 0x401f);
  return __builtin_bit_cast(float, i);
}

// Broadcast per-query stat (held in lanes q and q+16) to O's C-layout row r:
// lanes 0-15 need q=r (src lane r), lanes 16-31 need q=r+8 (src lane r+8).
__device__ __forceinline__ float bcast_row(float v, int r, int lane) {
  int src = (lane < 16) ? r : (r + 8);
  int i = __builtin_amdgcn_ds_bpermute(src << 2, __builtin_bit_cast(int, v));
  return __builtin_bit_cast(float, i);
}

// v8f -> 4 dwords of packed bf16 (vector fpround -> v_cvt_pk_bf16_f32 x4).
__device__ __forceinline__ uint4 cvt8_to_bf16(v8f f) {
  union { v8bf b; uint4 u; } c;
  c.b = __builtin_convertvector(f, v8bf);
  return c.u;
}

// Build a v16bf fragment from two 16B-aligned LDS dword quads
// (elements 0..7 from p0[0..3], 8..15 from p1[0..3]).
__device__ __forceinline__ v16bf frag_from_lds(const unsigned int* p0,
                                               const unsigned int* p1) {
  union { uint4 q[2]; v16bf v; } u;
  u.q[0] = *(const uint4*)p0;
  u.q[1] = *(const uint4*)p1;
  return u.v;
}

__global__ __launch_bounds__(NTHREADS)
void fa_causal_wmma_gfx1250(const float* __restrict__ Q,
                            const float* __restrict__ K,
                            const float* __restrict__ V,
                            float* __restrict__ Out) {
  __shared__ float        Qlds[BLOCK_M * kD];        // 32 KB f32 Q block
  __shared__ unsigned int KVlds[2][BLOCK_N * KVSTR]; // 2 x 4.5 KB packed-bf16 K / V

  const int tid  = threadIdx.x;
  const int wave = tid >> 5;
  const int lane = tid & 31;
  const int bh   = blockIdx.y;
  const int qblk = blockIdx.x;
  const size_t base = (size_t)bh * kS * kD;

  // --- cooperative Q block load to LDS (row-major f32) ---
  const float* Qg = Q + base + (size_t)qblk * BLOCK_M * kD;
  for (int i = tid; i < BLOCK_M * kD / 4; i += NTHREADS)
    ((float4*)Qlds)[i] = ((const float4*)Qg)[i];
  __syncthreads();

  // --- build Q^T B-fragments (64 x 16) from LDS ---
  // Scale folds 1/sqrt(dk) AND log2(e): softmax runs in the log2 domain so
  // exp() is a single v_exp_f32 (exp2) per element, numerically identical.
  const int q0 = wave * 16;
  const float scale = 0.125f * 1.44269504088896340736f;
  v16f q0f, q1f;
#pragma unroll
  for (int e = 0; e < 16; ++e) {
    q0f[e] = Qlds[(q0 + (lane & 15)) * kD + ((lane & 16) ? 8 : 0) + (e & 7) + ((e >> 3) << 4)];
    q1f[e] = Qlds[(q0 + (lane & 15)) * kD + 32 + ((lane & 16) ? 8 : 0) + (e & 7) + ((e >> 3) << 4)];
  }
  const v16bf bq0 = __builtin_convertvector(q0f * scale, v16bf);
  const v16bf bq1 = __builtin_convertvector(q1f * scale, v16bf);

  const int qbase = qblk * BLOCK_M + q0;       // wave's first query row
  const int qg    = qbase + (lane & 15);       // this lane's query row
  const int kblo  = (lane & 16) ? 8 : 0;       // A-frag K offset per lane half
  const float* Kb = K + base;
  const float* Vb = V + base;

  // staging role: threads 0..127 stage K, 128..255 stage V; 16 f32 each
  const int sHalf = tid >> 7;
  const int sRow  = (tid & 127) >> 2;          // 0..31 within block
  const int sCol  = (tid & 3) * 16;            // f32 column of 16-wide slice
  const float* SrcB = sHalf ? Vb : Kb;

  float m = -1e30f, lsum = 0.0f;               // m in log2 domain
  v8f o[4] = {};                               // O accum: 4 d-tiles of 16x16 f32

  const int nblkWG   = qblk * 4 + 4;           // workgroup causal extent (32-key blocks)
  const int nblkWave = (qbase + 15) / BLOCK_N + 1;

  // prologue: load block 0 into staging registers
  v8f rga, rgb;
  {
    const float* p = SrcB + (size_t)sRow * kD + sCol;
    rga = *(const v8f*)(p);
    rgb = *(const v8f*)(p + 8);
  }

  for (int jb = 0; jb < nblkWG; ++jb) {
    const int kb = jb * BLOCK_N;

    // --- stage current block (vector convert + pack to bf16 LDS) ---
    {
      unsigned int* d = &KVlds[sHalf][sRow * KVSTR + (sCol >> 1)];
      *(uint4*)(d)     = cvt8_to_bf16(rga);
      *(uint4*)(d + 4) = cvt8_to_bf16(rgb);
    }
    __syncthreads();

    // --- software pipeline: issue next block's global loads now ---
    if (jb + 1 < nblkWG) {
      const float* p = SrcB + (size_t)(kb + BLOCK_N + sRow) * kD + sCol;
      rga = *(const v8f*)(p);
      rgb = *(const v8f*)(p + 8);
    }

    if (jb < nblkWave) {
      const unsigned int* KL = KVlds[0];
      const unsigned int* VL = KVlds[1];

      // --- S^T = K_tile(16x64) @ Q^T(64x16) for two 16-key subtiles ---
      v8f s[2];
#pragma unroll
      for (int t = 0; t < 2; ++t) {
        const unsigned int* kr =
            KL + (t * 16 + (lane & 15)) * KVSTR + (kblo >> 1);
        v16bf ka0 = frag_from_lds(kr,      kr + 8);   // d 0..31
        v16bf ka1 = frag_from_lds(kr + 16, kr + 24);  // d 32..63
        v8f z = {0.f, 0.f, 0.f, 0.f, 0.f, 0.f, 0.f, 0.f};
        v8f c = WMMA_BF16(ka0, bq0, z);
        c     = WMMA_BF16(ka1, bq1, c);
        s[t] = c;
      }

      // --- causal mask only on diagonal blocks ---
      float mt = -1e30f;
      if (kb + BLOCK_N - 1 > qbase) {
#pragma unroll
        for (int t = 0; t < 2; ++t)
#pragma unroll
          for (int r = 0; r < 8; ++r) {
            int   kg = kb + t * 16 + kblo + r;
            float v  = (kg <= qg) ? s[t][r] : -1e30f;
            s[t][r]  = v;
            mt = fmaxf(mt, v);
          }
      } else {
#pragma unroll
        for (int t = 0; t < 2; ++t)
#pragma unroll
          for (int r = 0; r < 8; ++r) mt = fmaxf(mt, s[t][r]);
      }

      // --- online softmax, log2 domain (combine key halves via SWAPX16) ---
      mt = fmaxf(mt, xor16f(mt));
      float mnew  = fmaxf(m, mt);
      float alpha = __builtin_amdgcn_exp2f(m - mnew);
      float rs = 0.f;
#pragma unroll
      for (int t = 0; t < 2; ++t)
#pragma unroll
        for (int r = 0; r < 8; ++r) {
          float p = __builtin_amdgcn_exp2f(s[t][r] - mnew);
          s[t][r] = p;
          rs += p;
        }
      rs += xor16f(rs);
      lsum = lsum * alpha + rs;
      m = mnew;

      // --- S^T C-layout IS P's A-layout: one 16-wide vector fpround ---
      v16f sf = __builtin_shufflevector(s[0], s[1],
                                        0, 1, 2, 3, 4, 5, 6, 7,
                                        8, 9, 10, 11, 12, 13, 14, 15);
      const v16bf pa = __builtin_convertvector(sf, v16bf);

      // --- per-O-row alpha broadcast (C-layout row r <-> query r / r+8) ---
      float aB[8];
#pragma unroll
      for (int r = 0; r < 8; ++r) aB[r] = bcast_row(alpha, r, lane);

      // --- O = O*alpha + P(16x32) @ V(32x64); V B-frags from LDS ---
#pragma unroll
      for (int dt = 0; dt < 4; ++dt) {
        v16bf vb;
#pragma unroll
        for (int e = 0; e < 16; ++e)
          vb[e] = ((const __bf16*)(VL + (kblo + (e & 7) + ((e >> 3) << 4)) * KVSTR))[dt * 16 + (lane & 15)];
        v8f oc = o[dt];
#pragma unroll
        for (int r = 0; r < 8; ++r) oc[r] *= aB[r];
        o[dt] = WMMA_BF16(pa, vb, oc);
      }
    }

    __syncthreads();  // protect LDS before next iteration's staging store
  }

  // --- normalize by 1/l and store f32 (16-float coalesced runs per half-wave) ---
  float linv = 1.0f / lsum;
  float lB[8];
#pragma unroll
  for (int r = 0; r < 8; ++r) lB[r] = bcast_row(linv, r, lane);

  float* Ob = Out + base;
#pragma unroll
  for (int dt = 0; dt < 4; ++dt)
#pragma unroll
    for (int r = 0; r < 8; ++r) {
      int qrow = qbase + r + ((lane & 16) ? 8 : 0);
      Ob[(size_t)qrow * kD + dt * 16 + (lane & 15)] = o[dt][r] * lB[r];
    }
}

extern "C" void kernel_launch(void* const* d_in, const int* in_sizes, int n_in,
                              void* d_out, int out_size, void* d_ws, size_t ws_size,
                              hipStream_t stream) {
  (void)in_sizes; (void)n_in; (void)out_size; (void)d_ws; (void)ws_size;
  const float* Q = (const float*)d_in[0];
  const float* K = (const float*)d_in[1];
  const float* V = (const float*)d_in[2];
  float* O = (float*)d_out;
  dim3 grid(kS / BLOCK_M, kB * kH);
  dim3 block(NTHREADS);
  fa_causal_wmma_gfx1250<<<grid, block, 0, stream>>>(Q, K, V, O);
}
